// NodeScalarFeatInit_73452530696697
// MI455X (gfx1250) — matrix operand order repaired and checked by reading, not compiled, not run on hardware
//
#include <hip/hip_runtime.h>

#define BB 4
#define NN 4096
#define DD 128
#define WAVES 4
#define ROWS (WAVES * 16)
#define JCH 64                 // j-chunk per staged B tile (2 WMMA K-steps)
#define NSTAGE (NN / JCH)      // 64 stages
#define BT_PAD 72              // btile row stride: 144B=36 banks -> conflict-free
#define LN_EPS 1e-5f

typedef __bf16 bf16;
typedef __attribute__((ext_vector_type(8)))  __bf16 v8bf;
typedef __attribute__((ext_vector_type(16))) __bf16 v16bf;
typedef __attribute__((ext_vector_type(8)))  float  v8f;
typedef int v4i_b __attribute__((vector_size(16)));   // builtin's pointee type

__device__ __forceinline__ v16bf cat8(v8bf lo, v8bf hi) {
  return __builtin_shufflevector(lo, hi, 0,1,2,3,4,5,6,7,8,9,10,11,12,13,14,15);
}

__device__ __forceinline__ v8f wmma_bf16(v16bf a, v16bf b, v8f c) {
  return __builtin_amdgcn_wmma_f32_16x16x32_bf16(false, a, false, b, (short)0, c,
                                                 false, false);
}

// ---- CDNA5 async global->LDS copy (ASYNCcnt-tracked), with safe fallback ----
#if __has_builtin(__builtin_amdgcn_global_load_async_to_lds_b128)
#define HAVE_ASYNC 1
__device__ __forceinline__ void async_cp16(const void* g, void* l) {
  __builtin_amdgcn_global_load_async_to_lds_b128(
      (__attribute__((address_space(1))) v4i_b*)(g),
      (__attribute__((address_space(3))) v4i_b*)(l), 0, 0);
}
__device__ __forceinline__ void async_wait_le8() {
#if __has_builtin(__builtin_amdgcn_s_wait_asynccnt)
  __builtin_amdgcn_s_wait_asynccnt(8);
#else
  asm volatile("s_wait_asynccnt 0x8" ::: "memory");
#endif
}
__device__ __forceinline__ void async_wait_0() {
#if __has_builtin(__builtin_amdgcn_s_wait_asynccnt)
  __builtin_amdgcn_s_wait_asynccnt(0);
#else
  asm volatile("s_wait_asynccnt 0x0" ::: "memory");
#endif
}
#else
#define HAVE_ASYNC 0
__device__ __forceinline__ void async_cp16(const void* g, void* l) {
  *(v8bf*)l = *(const v8bf*)g;     // load + ds_store fallback
}
__device__ __forceinline__ void async_wait_le8() {}
__device__ __forceinline__ void async_wait_0() {}
#endif

// ---------------------------------------------------------------------------
// Prep kernels: bf16 gathers / transposes so the hot kernel streams cleanly.
// ---------------------------------------------------------------------------
__global__ __launch_bounds__(256)
void prep_embed_kernel(const int* __restrict__ ids, const float* __restrict__ tab,
                       bf16* __restrict__ out) {
  const size_t total = (size_t)BB * NN * (DD / 4);
  const size_t idx = (size_t)blockIdx.x * blockDim.x + threadIdx.x;
  if (idx >= total) return;
  const size_t bn = idx / (DD / 4);
  const int d4 = (int)(idx % (DD / 4));
  const int id = ids[bn];
  const float4 v = *(const float4*)(tab + (size_t)id * DD + d4 * 4);
  bf16* o = out + bn * DD + d4 * 4;
  o[0] = (bf16)v.x; o[1] = (bf16)v.y; o[2] = (bf16)v.z; o[3] = (bf16)v.w;
}

__global__ __launch_bounds__(256)
void prep_nbT_kernel(const int* __restrict__ ids, const float* __restrict__ tab,
                     bf16* __restrict__ out) {
  const size_t total = (size_t)BB * DD * NN;
  const size_t idx = (size_t)blockIdx.x * blockDim.x + threadIdx.x;
  if (idx >= total) return;
  const int j = (int)(idx % NN);
  const size_t bd = idx / NN;
  const int d = (int)(bd % DD);
  const int b = (int)(bd / DD);
  const int id = ids[(size_t)b * NN + j];
  out[idx] = (bf16)tab[(size_t)id * DD + d];
}

__global__ __launch_bounds__(256)
void prep_w_kernel(const float* __restrict__ w1, const float* __restrict__ w2,
                   bf16* __restrict__ w1T, bf16* __restrict__ w2T) {
  const int idx = blockIdx.x * 256 + threadIdx.x;
  if (idx < 128 * 256) {
    const int n = idx / 256, k = idx % 256;
    w1T[idx] = (bf16)w1[k * 128 + n];
  } else if (idx < 128 * 256 + 128 * 128) {
    const int i2 = idx - 128 * 256;
    const int n = i2 / 128, k = i2 % 128;
    w2T[i2] = (bf16)w2[k * 128 + n];
  }
}

// ---------------------------------------------------------------------------
// Fused kernel. Each wave owns 16 rows x 128 cols (8 WMMA C-tiles).
// B tiles staged global->LDS via async copy, triple-buffered, 1 barrier/stage.
// ---------------------------------------------------------------------------
__global__ __launch_bounds__(WAVES * 32)
void fused_gnn_kernel(const unsigned char* __restrict__ adj,
                      const float* __restrict__ rel,
                      const float* __restrict__ cut,
                      const bf16* __restrict__ embq,   // [B,N,D]
                      const bf16* __restrict__ nbT,    // [B,D,N]
                      const bf16* __restrict__ w1T,    // [128,256]
                      const bf16* __restrict__ w2T,    // [128,128]
                      const float* __restrict__ rdw,
                      const float* __restrict__ b1,
                      const float* __restrict__ ln_g,
                      const float* __restrict__ ln_b,
                      const float* __restrict__ b2,
                      float* __restrict__ out) {
  __shared__ bf16 btile[3][DD][BT_PAD];    // 3 x 18KB staged B tiles
  __shared__ bf16 stage[WAVES][16][136];   // per-wave activation staging

  const int tid  = threadIdx.x;
  const int wave = tid >> 5;
  const int lane = tid & 31;
  const int hf   = lane >> 4;     // 16-lane half
  const int lr   = lane & 15;     // A-row / B,C-column within tile

  const int blocksPerBatch = NN / ROWS;
  const int b   = blockIdx.x / blocksPerBatch;
  const int i0  = (blockIdx.x % blocksPerBatch) * ROWS + wave * 16;
  const int row = i0 + lr;        // global row this lane feeds into A

  const v8f vzero = {0.f,0.f,0.f,0.f,0.f,0.f,0.f,0.f};

  const size_t rowOff = ((size_t)b * NN + row) * NN;
  const float* relRow = rel + rowOff;
  const float* cutRow = cut + rowOff;
  const unsigned char* adjRow = adj + rowOff;
  const bf16* nbTb = nbT + (size_t)b * DD * NN;

  // ============ GEMM1: nf[16,128] = (mask*rel*cutoff)[16,4096] @ nbT^T =====
  v8f acc[8];
  #pragma unroll
  for (int t = 0; t < 8; ++t) acc[t] = vzero;

  // whole block cooperatively stages one [128 x 64] bf16 B tile: thread tid
  // copies row d=tid (128B) as 8 x 16B async transfers.
  {
    bf16* l = &btile[0][0][0] + (size_t)tid * BT_PAD;
    const bf16* g = nbTb + (size_t)tid * NN;
    #pragma unroll
    for (int k = 0; k < 8; ++k) async_cp16(g + 8 * k, l + 8 * k);
  }

  for (int s = 0; s < NSTAGE; ++s) {
    const int j0 = s * JCH;
    const int buf = s % 3;
    if (s + 1 < NSTAGE) {                      // prefetch next stage's tile
      bf16* l = &btile[(s + 1) % 3][0][0] + (size_t)tid * BT_PAD;
      const bf16* g = nbTb + (size_t)tid * NN + (j0 + JCH);
      #pragma unroll
      for (int k = 0; k < 8; ++k) async_cp16(g + 8 * k, l + 8 * k);
      async_wait_le8();                        // in-order: stage s tile landed
    } else {
      async_wait_0();
    }
    __syncthreads();

    __builtin_prefetch(relRow + j0 + 2 * JCH, 0, 1);   // global_prefetch_b8
    __builtin_prefetch(cutRow + j0 + 2 * JCH, 0, 1);

    #pragma unroll
    for (int ks = 0; ks < 2; ++ks) {
      const int jb = j0 + 32 * ks;
      const int k0 = jb + 16 * hf;             // A K-run 0 (8 contiguous j)
      const int k1 = jb + 16 * hf + 8;         // A K-run 1
      const float4 r0a = *(const float4*)(relRow + k0);
      const float4 r0b = *(const float4*)(relRow + k0 + 4);
      const float4 r1a = *(const float4*)(relRow + k1);
      const float4 r1b = *(const float4*)(relRow + k1 + 4);
      const float4 c0a = *(const float4*)(cutRow + k0);
      const float4 c0b = *(const float4*)(cutRow + k0 + 4);
      const float4 c1a = *(const float4*)(cutRow + k1);
      const float4 c1b = *(const float4*)(cutRow + k1 + 4);
      const unsigned long long m0 = *(const unsigned long long*)(adjRow + k0);
      const unsigned long long m1 = *(const unsigned long long*)(adjRow + k1);

      const float r0[8] = {r0a.x,r0a.y,r0a.z,r0a.w,r0b.x,r0b.y,r0b.z,r0b.w};
      const float r1[8] = {r1a.x,r1a.y,r1a.z,r1a.w,r1b.x,r1b.y,r1b.z,r1b.w};
      const float c0[8] = {c0a.x,c0a.y,c0a.z,c0a.w,c0b.x,c0b.y,c0b.z,c0b.w};
      const float c1[8] = {c1a.x,c1a.y,c1a.z,c1a.w,c1b.x,c1b.y,c1b.z,c1b.w};

      v16bf a;
      #pragma unroll
      for (int e = 0; e < 8; ++e) {
        const bool on0 = (((m0 >> (8 * e)) & 0xffull) != 0) && (k0 + e != row);
        const bool on1 = (((m1 >> (8 * e)) & 0xffull) != 0) && (k1 + e != row);
        a[e]     = (bf16)(on0 ? r0[e] * c0[e] : 0.f);
        a[8 + e] = (bf16)(on1 ? r1[e] * c1[e] : 0.f);
      }
      #pragma unroll
      for (int t = 0; t < 8; ++t) {
        const bf16* bp = &btile[buf][16 * t + lr][32 * ks + 16 * hf];
        const v16bf bfrag = cat8(*(const v8bf*)bp, *(const v8bf*)(bp + 8));
        acc[t] = wmma_bf16(a, bfrag, acc[t]);
      }
    }
  }

  // scale by rel_dist_w[d]; stash bf16 row-major for A-fragment reload
  #pragma unroll
  for (int t = 0; t < 8; ++t) {
    const float s = rdw[16 * t + lr];
    const v8f v = acc[t] * s;
    #pragma unroll
    for (int m = 0; m < 8; ++m)
      stage[wave][m + 8 * hf][16 * t + lr] = (bf16)v[m];
  }

  // ============ GEMM2: h = [emb, nf] @ w1  (K = 256) =======================
  v8f h[8];
  #pragma unroll
  for (int t = 0; t < 8; ++t) h[t] = vzero;

  const bf16* embRow = embq + ((size_t)b * NN + row) * DD;
  #pragma unroll
  for (int c = 0; c < 4; ++c) {          // K chunks 0..127: embeds half
    const v8bf lo = *(const v8bf*)(embRow + 32 * c + 16 * hf);
    const v8bf hi = *(const v8bf*)(embRow + 32 * c + 16 * hf + 8);
    const v16bf a = cat8(lo, hi);
    #pragma unroll
    for (int t = 0; t < 8; ++t) {
      const v16bf bfrag =
          *(const v16bf*)(w1T + (size_t)(16 * t + lr) * 256 + 32 * c + 16 * hf);
      h[t] = wmma_bf16(a, bfrag, h[t]);
    }
  }
  #pragma unroll
  for (int c = 0; c < 4; ++c) {          // K chunks 128..255: nf half (LDS)
    const v8bf lo = *(const v8bf*)&stage[wave][lr][32 * c + 16 * hf];
    const v8bf hi = *(const v8bf*)&stage[wave][lr][32 * c + 16 * hf + 8];
    const v16bf a = cat8(lo, hi);
    #pragma unroll
    for (int t = 0; t < 8; ++t) {
      const v16bf bfrag =
          *(const v16bf*)(w1T + (size_t)(16 * t + lr) * 256 + 128 + 32 * c + 16 * hf);
      h[t] = wmma_bf16(a, bfrag, h[t]);
    }
  }

  float gv[8], bv[8], b2v[8];
  #pragma unroll
  for (int t = 0; t < 8; ++t) {
    const int n = 16 * t + lr;
    gv[t] = ln_g[n]; bv[t] = ln_b[n]; b2v[t] = b2[n];
    h[t] += b1[n];
  }

  // ============ LayerNorm + SiLU (row = vgpr slot; halves hold rows) =======
  #pragma unroll
  for (int m = 0; m < 8; ++m) {
    float s = 0.f;
    #pragma unroll
    for (int t = 0; t < 8; ++t) s += h[t][m];
    s += __shfl_xor(s, 1, 32); s += __shfl_xor(s, 2, 32);
    s += __shfl_xor(s, 4, 32); s += __shfl_xor(s, 8, 32);
    const float mu = s * (1.f / 128.f);
    float q = 0.f;
    #pragma unroll
    for (int t = 0; t < 8; ++t) { const float d = h[t][m] - mu; q += d * d; }
    q += __shfl_xor(q, 1, 32); q += __shfl_xor(q, 2, 32);
    q += __shfl_xor(q, 4, 32); q += __shfl_xor(q, 8, 32);
    const float rs = rsqrtf(q * (1.f / 128.f) + LN_EPS);
    #pragma unroll
    for (int t = 0; t < 8; ++t) {
      float x = (h[t][m] - mu) * rs * gv[t] + bv[t];
      x = x / (1.f + __expf(-x));        // SiLU
      h[t][m] = x;
    }
  }

  #pragma unroll
  for (int t = 0; t < 8; ++t)
    #pragma unroll
    for (int m = 0; m < 8; ++m)
      stage[wave][m + 8 * hf][16 * t + lr] = (bf16)h[t][m];

  // ============ GEMM3: out = h @ w2 + b2  (K = 128) ========================
  v8f o[8];
  #pragma unroll
  for (int t = 0; t < 8; ++t) o[t] = vzero;
  #pragma unroll
  for (int c = 0; c < 4; ++c) {
    const v8bf lo = *(const v8bf*)&stage[wave][lr][32 * c + 16 * hf];
    const v8bf hi = *(const v8bf*)&stage[wave][lr][32 * c + 16 * hf + 8];
    const v16bf a = cat8(lo, hi);
    #pragma unroll
    for (int t = 0; t < 8; ++t) {
      const v16bf bfrag =
          *(const v16bf*)(w2T + (size_t)(16 * t + lr) * 128 + 32 * c + 16 * hf);
      o[t] = wmma_bf16(a, bfrag, o[t]);
    }
  }
  #pragma unroll
  for (int t = 0; t < 8; ++t)
    #pragma unroll
    for (int m = 0; m < 8; ++m)
      out[((size_t)b * NN + i0 + m + 8 * hf) * DD + 16 * t + lr] = o[t][m] + b2v[t];
}

// ---------------------------------------------------------------------------
extern "C" void kernel_launch(void* const* d_in, const int* in_sizes, int n_in,
                              void* d_out, int out_size, void* d_ws, size_t ws_size,
                              hipStream_t stream) {
  (void)in_sizes; (void)n_in; (void)out_size; (void)ws_size;
  const int* atom_ids            = (const int*)d_in[0];
  const unsigned char* adj       = (const unsigned char*)d_in[1];
  const float* rel               = (const float*)d_in[2];
  const float* cut               = (const float*)d_in[3];
  const float* atom_embed        = (const float*)d_in[4];
  const float* nb_embed          = (const float*)d_in[5];
  const float* rdw               = (const float*)d_in[6];
  const float* w1                = (const float*)d_in[7];
  const float* b1                = (const float*)d_in[8];
  const float* ln_g              = (const float*)d_in[9];
  const float* ln_b              = (const float*)d_in[10];
  const float* w2                = (const float*)d_in[11];
  const float* b2                = (const float*)d_in[12];
  float* out = (float*)d_out;

  bf16* emb = (bf16*)d_ws;                      // [B,N,D]   4 MB
  bf16* nbT = emb + (size_t)BB * NN * DD;       // [B,D,N]   4 MB
  bf16* w1T = nbT + (size_t)BB * NN * DD;       // [128,256] 64 KB
  bf16* w2T = w1T + 256 * 128;                  // [128,128] 32 KB

  prep_embed_kernel<<<(BB * NN * (DD / 4) + 255) / 256, 256, 0, stream>>>(
      atom_ids, atom_embed, emb);
  prep_nbT_kernel<<<(BB * DD * NN + 255) / 256, 256, 0, stream>>>(
      atom_ids, nb_embed, nbT);
  prep_w_kernel<<<(128 * 256 + 128 * 128 + 255) / 256, 256, 0, stream>>>(
      w1, w2, w1T, w2T);
  fused_gnn_kernel<<<BB * (NN / ROWS), WAVES * 32, 0, stream>>>(
      adj, rel, cut, emb, nbT, w1T, w2T, rdw, b1, ln_g, ln_b, b2, out);
}
